// Attention_30004641529862
// MI455X (gfx1250) — matrix operand order repaired and checked by reading, not compiled, not run on hardware
//
#include <hip/hip_runtime.h>

// ---------------- problem constants (match reference) ----------------
#define NUM_HEADS     32
#define NUM_KV_HEADS  8
#define HEAD_DIM      128
#define GQA           4
#define SCALE         0.08838834764831845f
#define NUM_SEQS      4
#define SEQLEN        1024
#define N_PREFILL     4096
#define DECODE_BATCH  32
#define NUM_BLOCKS    256
#define BLOCK_SIZE    256
#define MAX_BLOCKS    8
#define MAX_SEQLEN_K  2048
#define TOTAL         4128
#define NEG_INF       -1e30f
#define NUM_SLOTS     (NUM_BLOCKS * BLOCK_SIZE)   // 65536
#define KVSTRIDE      (NUM_KV_HEADS * HEAD_DIM)   // 1024

// ---------------- CDNA5 WMMA types ----------------
typedef __bf16 bf16_t;
typedef bf16_t bfvec16 __attribute__((ext_vector_type(16)));
typedef float  fvec8   __attribute__((ext_vector_type(8)));
typedef unsigned uvec4 __attribute__((ext_vector_type(4)));
typedef int      ivec8 __attribute__((ext_vector_type(8)));
typedef int      ivec4 __attribute__((ext_vector_type(4)));

union FragU {           // assemble bf16 fragments from packed 32-bit pairs
    bfvec16  v;
    unsigned u[8];
};

__device__ __forceinline__ fvec8 wmma_bf16(bfvec16 a, bfvec16 b, fvec8 c) {
    return __builtin_amdgcn_wmma_f32_16x16x32_bf16(false, a, false, b,
                                                   (short)0, c, false, false);
}

// fp32 -> bf16 bits (truncate; 1 shift)
__device__ __forceinline__ unsigned short f2bf(float f) {
    return (unsigned short)(__builtin_bit_cast(unsigned, f) >> 16);
}
// two fp32 -> packed bf16x2: v_cvt_pk if present, else single v_perm_b32
__device__ __forceinline__ unsigned pack_bf16(float x, float y) {
#if __has_builtin(__builtin_amdgcn_cvt_pk_bf16_f32)
    auto r = __builtin_amdgcn_cvt_pk_bf16_f32(x, y);
    return __builtin_bit_cast(unsigned, r);
#else
    return __builtin_amdgcn_perm(__builtin_bit_cast(unsigned, y),
                                 __builtin_bit_cast(unsigned, x), 0x07060302u);
#endif
}
__device__ __forceinline__ bf16_t bfbits(unsigned short u) {
    return __builtin_bit_cast(bf16_t, u);
}

// A-fragment (16-bit, 16x32) K index for lane l, element e (contiguous: k = 16*(l/16) + e)
__device__ __forceinline__ int a_kidx(int l, int e) {
    return ((l >> 4) << 4) | e;
}

__device__ __forceinline__ bfvec16 ones_frag() {
    FragU f;
#pragma unroll
    for (int j = 0; j < 8; ++j) f.u[j] = 0x3F803F80u;  // bf16 1.0 pair
    return f.v;
}

// ---------------- TDM: 2D tile (32 keys x 128 f32, row stride 1024) -> LDS ----
#if __has_builtin(__builtin_amdgcn_tensor_load_to_lds)
#define HAVE_TDM 1
__device__ __forceinline__ void tdm_load_k(const float* gptr, unsigned lds_off) {
    unsigned long long ga = (unsigned long long)(uintptr_t)gptr;
    uvec4 g0;
    g0[0] = 1u;                                            // count=1, user D#
    g0[1] = lds_off;                                       // lds_addr (bytes)
    g0[2] = (unsigned)(ga & 0xFFFFFFFFu);                  // global_addr[31:0]
    g0[3] = (unsigned)((ga >> 32) & 0x01FFFFFFu) | (2u << 30); // addr[56:32]|type=2
    ivec8 g1;
    g1[0] = 0x20000;                 // workgroup_mask=0 | data_size=2 (4B)
    g1[1] = (int)(128u << 16);       // tensor_dim0 = 128 (low 16 of bits 79:48)
    g1[2] = (int)(32u << 16);        // tensor_dim0 hi | tensor_dim1 = 32
    g1[3] = (int)(128u << 16);       // tensor_dim1 hi | tile_dim0 = 128
    g1[4] = 32;                      // tile_dim1 = 32 | tile_dim2 = 0
    g1[5] = KVSTRIDE;                // tensor_dim0_stride = 1024 (elem units)
    g1[6] = 0; g1[7] = 0;
    ivec4 z4 = {0, 0, 0, 0};
    ivec8 z8 = {0, 0, 0, 0, 0, 0, 0, 0};
    // clang-23 / therock arity: (g0, g1, g2, g3, g4, cpol)
    __builtin_amdgcn_tensor_load_to_lds(g0, g1, z4, z4, z8, 0);
}
#else
#define HAVE_TDM 0
#endif

// ---------------- inverse slot table ----------------
__global__ void init_inv_kernel(int* inv) {
    int i = blockIdx.x * blockDim.x + threadIdx.x;
    if (i < NUM_SLOTS) inv[i] = -1;
}
__global__ void scatter_inv_kernel(const long long* slot_mapping, int* inv) {
    int i = blockIdx.x * blockDim.x + threadIdx.x;
    if (i < TOTAL) inv[(int)slot_mapping[i]] = i;
}

// ---------------- prefill flash attention ----------------
// grid: NUM_SEQS*NUM_HEADS*(SEQLEN/64) blocks, 128 threads (4 waves).
// Double-buffered: TDM streams K tiles (f32), threads stage V^T (bf16).
__global__ void __launch_bounds__(128)
prefill_fa_kernel(const float* __restrict__ q, const float* __restrict__ k,
                  const float* __restrict__ v, float* __restrict__ out) {
    __shared__ float          sKf[2][32 * HEAD_DIM];   // K tiles, f32, 2x16KB
    __shared__ unsigned short sVt[2][HEAD_DIM * 32];   // V^T tiles, bf16, 2x8KB
    __shared__ unsigned short sP[4][16 * 32];          // per-wave P, 4KB

    const int bid = blockIdx.x;
    const int qb  = bid & 15;
    const int h   = (bid >> 4) & 31;
    const int seq = bid >> 9;
    const int kvh = h >> 2;
    const int q0  = qb * 64;

    const int tid  = threadIdx.x;
    const int wid  = tid >> 5;
    const int lane = tid & 31;
    const int ln16 = lane & 15;
    const int rowb = (lane >> 4) << 3;

    const float* kbaseb = k + (size_t)seq * SEQLEN * KVSTRIDE + (size_t)kvh * HEAD_DIM;
    const float* vbaseb = v + (size_t)seq * SEQLEN * KVSTRIDE + (size_t)kvh * HEAD_DIM;

    // ---- Q fragments (A layout), pre-scaled, packed bf16 ----
    const int   qrow = q0 + wid * 16 + ln16;
    const float* qp  = q + ((size_t)(seq * SEQLEN + qrow) * NUM_HEADS + h) * HEAD_DIM;
    bfvec16 qf[4];
#pragma unroll
    for (int c = 0; c < 4; ++c) {
        FragU f;
#pragma unroll
        for (int j = 0; j < 8; ++j) {
            const int d = c * 32 + a_kidx(lane, 2 * j);
            f.u[j] = pack_bf16(qp[d] * SCALE, qp[d + 1] * SCALE);
        }
        qf[c] = f.v;
    }

    const bfvec16 ones = ones_frag();
    fvec8 o[8];
    float m_i[8], l_i[8];
#pragma unroll
    for (int c = 0; c < 8; ++c) o[c] = fvec8{0, 0, 0, 0, 0, 0, 0, 0};
#pragma unroll
    for (int r = 0; r < 8; ++r) { m_i[r] = NEG_INF; l_i[r] = 0.f; }

    const int ntiles = 2 * qb + 2;
    const int kp2 = (tid & 15) * 2;   // V staging: key pair
    const int d0  = (tid >> 4) * 16;  // V staging: 16 dims

    // ---- prologue: stage tile 0 into buffer 0 ----
    float4 rg[8];
    {
        const float* v0 = vbaseb + (size_t)kp2 * KVSTRIDE;
#pragma unroll
        for (int i = 0; i < 4; ++i) {
            rg[2 * i]     = *(const float4*)(v0 + d0 + 4 * i);
            rg[2 * i + 1] = *(const float4*)(v0 + KVSTRIDE + d0 + 4 * i);
        }
#if HAVE_TDM
        if (wid == 0) tdm_load_k(kbaseb, (unsigned)(size_t)&sKf[0][0]);
#else
        {
            const int key = tid >> 2, dd = (tid & 3) * 32;
            const float* kp0 = kbaseb + (size_t)key * KVSTRIDE;
#pragma unroll
            for (int d = 0; d < 32; d += 4)
                *(float4*)&sKf[0][key * HEAD_DIM + dd + d] = *(const float4*)(kp0 + dd + d);
        }
#endif
#pragma unroll
        for (int i = 0; i < 4; ++i) {
            const int d = d0 + 4 * i;
            *(unsigned*)&sVt[0][(d + 0) * 32 + kp2] = pack_bf16(rg[2 * i].x, rg[2 * i + 1].x);
            *(unsigned*)&sVt[0][(d + 1) * 32 + kp2] = pack_bf16(rg[2 * i].y, rg[2 * i + 1].y);
            *(unsigned*)&sVt[0][(d + 2) * 32 + kp2] = pack_bf16(rg[2 * i].z, rg[2 * i + 1].z);
            *(unsigned*)&sVt[0][(d + 3) * 32 + kp2] = pack_bf16(rg[2 * i].w, rg[2 * i + 1].w);
        }
#if HAVE_TDM
        if (wid == 0) __builtin_amdgcn_s_wait_tensorcnt(0);
#endif
    }
    __syncthreads();

    for (int kt = 0; kt < ntiles; ++kt) {
        const int cur  = kt & 1;
        const int kbase = kt * 32;
        const bool more = (kt + 1 < ntiles);

        // ---- issue next tile's loads early (TDM for K, VMEM for V) ----
        if (more) {
            const float* v0 = vbaseb + (size_t)(kbase + 32 + kp2) * KVSTRIDE;
#pragma unroll
            for (int i = 0; i < 4; ++i) {
                rg[2 * i]     = *(const float4*)(v0 + d0 + 4 * i);
                rg[2 * i + 1] = *(const float4*)(v0 + KVSTRIDE + d0 + 4 * i);
            }
#if HAVE_TDM
            if (wid == 0)
                tdm_load_k(kbaseb + (size_t)(kbase + 32) * KVSTRIDE,
                           (unsigned)(size_t)&sKf[cur ^ 1][0]);
#endif
        }

        // ---- K B-fragments from f32 LDS (packed converts), dense WMMA ----
        bfvec16 bk[8];
#pragma unroll
        for (int c = 0; c < 4; ++c) {
            const int dimb  = c * 32 + ((lane >> 4) << 4);
            const int base0 = ln16 * HEAD_DIM + dimb;
            const int base1 = (16 + ln16) * HEAD_DIM + dimb;
            FragU f0, f1;
#pragma unroll
            for (int j = 0; j < 8; ++j) {
                f0.u[j] = pack_bf16(sKf[cur][base0 + 2 * j], sKf[cur][base0 + 2 * j + 1]);
                f1.u[j] = pack_bf16(sKf[cur][base1 + 2 * j], sKf[cur][base1 + 2 * j + 1]);
            }
            bk[2 * c] = f0.v; bk[2 * c + 1] = f1.v;
        }
        __builtin_amdgcn_sched_barrier(0);
        fvec8 s0 = fvec8{0, 0, 0, 0, 0, 0, 0, 0};
        fvec8 s1 = fvec8{0, 0, 0, 0, 0, 0, 0, 0};
#pragma unroll
        for (int c = 0; c < 4; ++c) {
            s0 = wmma_bf16(qf[c], bk[2 * c + 0], s0);
            s1 = wmma_bf16(qf[c], bk[2 * c + 1], s1);
        }
        __builtin_amdgcn_sched_barrier(0);

        // ---- online softmax ----
        float mx[8];
#pragma unroll
        for (int r = 0; r < 8; ++r) {
            const int qg = q0 + wid * 16 + rowb + r;
            float a = s0[r], b = s1[r];
            if (kbase + ln16 > qg)      a = NEG_INF;
            if (kbase + 16 + ln16 > qg) b = NEG_INF;
            s0[r] = a; s1[r] = b;
            mx[r] = fmaxf(a, b);
        }
#pragma unroll
        for (int off = 1; off <= 8; off <<= 1)
#pragma unroll
            for (int r = 0; r < 8; ++r)
                mx[r] = fmaxf(mx[r], __shfl_xor(mx[r], off, 32));

        float alpha[8];
#pragma unroll
        for (int r = 0; r < 8; ++r) {
            const float mn = fmaxf(m_i[r], mx[r]);
            alpha[r] = __expf(m_i[r] - mn);
            m_i[r]   = mn;
            const float p0 = __expf(s0[r] - mn);
            const float p1 = __expf(s1[r] - mn);
            sP[wid][(rowb + r) * 32 + ln16]      = f2bf(p0);
            sP[wid][(rowb + r) * 32 + 16 + ln16] = f2bf(p1);
#pragma unroll
            for (int c = 0; c < 8; ++c) o[c][r] *= alpha[r];
        }
        asm volatile("s_wait_dscnt 0x0" ::: "memory");

        // ---- P fragment + V fragments, dense WMMA clause ----
        bfvec16 pf;
#pragma unroll
        for (int e = 0; e < 16; ++e)
            pf[e] = bfbits(sP[wid][ln16 * 32 + a_kidx(lane, e)]);
        bfvec16 vf[8];
#pragma unroll
        for (int c = 0; c < 8; ++c) {
            const int dimc = c * 16 + ln16;
            const int keyb = (lane >> 4) << 4;
#pragma unroll
            for (int e = 0; e < 16; ++e)
                vf[c][e] = bfbits(sVt[cur][dimc * 32 + keyb + e]);
        }
        __builtin_amdgcn_sched_barrier(0);
        fvec8 lf = wmma_bf16(pf, ones, fvec8{0, 0, 0, 0, 0, 0, 0, 0});
#pragma unroll
        for (int c = 0; c < 8; ++c)
            o[c] = wmma_bf16(pf, vf[c], o[c]);
        __builtin_amdgcn_sched_barrier(0);
#pragma unroll
        for (int r = 0; r < 8; ++r) l_i[r] = l_i[r] * alpha[r] + lf[r];

        // ---- finish next tile's staging, one barrier per iteration ----
        if (more) {
#pragma unroll
            for (int i = 0; i < 4; ++i) {
                const int d = d0 + 4 * i;
                unsigned short* dst = sVt[cur ^ 1];
                *(unsigned*)&dst[(d + 0) * 32 + kp2] = pack_bf16(rg[2 * i].x, rg[2 * i + 1].x);
                *(unsigned*)&dst[(d + 1) * 32 + kp2] = pack_bf16(rg[2 * i].y, rg[2 * i + 1].y);
                *(unsigned*)&dst[(d + 2) * 32 + kp2] = pack_bf16(rg[2 * i].z, rg[2 * i + 1].z);
                *(unsigned*)&dst[(d + 3) * 32 + kp2] = pack_bf16(rg[2 * i].w, rg[2 * i + 1].w);
            }
#if !HAVE_TDM
            const int key = tid >> 2, dd = (tid & 3) * 32;
            const float* kp0 = kbaseb + (size_t)(kbase + 32 + key) * KVSTRIDE;
#pragma unroll
            for (int d = 0; d < 32; d += 4)
                *(float4*)&sKf[cur ^ 1][key * HEAD_DIM + dd + d] = *(const float4*)(kp0 + dd + d);
#endif
        }
#if HAVE_TDM
        if (wid == 0) __builtin_amdgcn_s_wait_tensorcnt(0);
#endif
        __syncthreads();
    }

    // ---- epilogue ----
#pragma unroll
    for (int c = 0; c < 8; ++c)
#pragma unroll
        for (int r = 0; r < 8; ++r) {
            const int token = seq * SEQLEN + q0 + wid * 16 + rowb + r;
            out[((size_t)token * NUM_HEADS + h) * HEAD_DIM + c * 16 + ln16] =
                o[c][r] / l_i[r];
        }
}

// ---------------- paged GQA decode ----------------
__global__ void __launch_bounds__(32)
decode_fa_kernel(const float* __restrict__ q, const float* __restrict__ k,
                 const float* __restrict__ v, const float* __restrict__ k_cache,
                 const float* __restrict__ v_cache, const int* __restrict__ inv,
                 const int* __restrict__ context_lens,
                 const int* __restrict__ block_tables, float* __restrict__ out) {
    __shared__ unsigned short sK[32 * HEAD_DIM];
    __shared__ unsigned short sVt[HEAD_DIM * 32];
    __shared__ unsigned short sP[16 * 32];

    const int b    = blockIdx.x >> 3;
    const int kvh  = blockIdx.x & 7;
    const int lane = threadIdx.x;
    const int ln16 = lane & 15;
    const int rowb = (lane >> 4) << 3;
    const int ctx  = context_lens[b];
    const int token = N_PREFILL + b;

    bfvec16 qf[4];
    {
        const int head = kvh * GQA + (ln16 & 3);
        const float* qp = q + ((size_t)token * NUM_HEADS + head) * HEAD_DIM;
#pragma unroll
        for (int c = 0; c < 4; ++c) {
            FragU f;
#pragma unroll
            for (int j = 0; j < 8; ++j) {
                const int d = c * 32 + a_kidx(lane, 2 * j);
                f.u[j] = pack_bf16(qp[d] * SCALE, qp[d + 1] * SCALE);
            }
            qf[c] = f.v;
        }
    }

    const bfvec16 ones = ones_frag();
    fvec8 o[8];
    float m_i[8], l_i[8];
#pragma unroll
    for (int c = 0; c < 8; ++c) o[c] = fvec8{0, 0, 0, 0, 0, 0, 0, 0};
#pragma unroll
    for (int r = 0; r < 8; ++r) { m_i[r] = NEG_INF; l_i[r] = 0.f; }

    const int ntiles = (ctx + 31) >> 5;
    for (int kt = 0; kt < ntiles; ++kt) {
        const int kbase = kt * 32;
        {
            const int kp2 = (lane & 15) * 2;
            const int dh  = (lane >> 4) * 64;
            const float *ks0 = nullptr, *vs0 = nullptr, *ks1 = nullptr, *vs1 = nullptr;
#pragma unroll
            for (int j = 0; j < 2; ++j) {
                const int s = kbase + kp2 + j;
                const float **kp = j ? &ks1 : &ks0;
                const float **vp = j ? &vs1 : &vs0;
                if (s < ctx) {
                    const int blk  = block_tables[b * MAX_BLOCKS + (s >> 8)];
                    const int slot = blk * BLOCK_SIZE + (s & 255);
                    const int t    = inv[slot];
                    if (t >= 0) {
                        *kp = k + ((size_t)t * NUM_KV_HEADS + kvh) * HEAD_DIM;
                        *vp = v + ((size_t)t * NUM_KV_HEADS + kvh) * HEAD_DIM;
                    } else {
                        *kp = k_cache + ((size_t)slot * NUM_KV_HEADS + kvh) * HEAD_DIM;
                        *vp = v_cache + ((size_t)slot * NUM_KV_HEADS + kvh) * HEAD_DIM;
                    }
                }
            }
#pragma unroll
            for (int d = 0; d < 64; d += 4) {
                float4 ka = ks0 ? *(const float4*)(ks0 + dh + d) : float4{0, 0, 0, 0};
                float4 kb = ks1 ? *(const float4*)(ks1 + dh + d) : float4{0, 0, 0, 0};
                float4 va = vs0 ? *(const float4*)(vs0 + dh + d) : float4{0, 0, 0, 0};
                float4 vb = vs1 ? *(const float4*)(vs1 + dh + d) : float4{0, 0, 0, 0};
                *(unsigned*)&sK[(kp2 + 0) * HEAD_DIM + dh + d]     = pack_bf16(ka.x, ka.y);
                *(unsigned*)&sK[(kp2 + 0) * HEAD_DIM + dh + d + 2] = pack_bf16(ka.z, ka.w);
                *(unsigned*)&sK[(kp2 + 1) * HEAD_DIM + dh + d]     = pack_bf16(kb.x, kb.y);
                *(unsigned*)&sK[(kp2 + 1) * HEAD_DIM + dh + d + 2] = pack_bf16(kb.z, kb.w);
                *(unsigned*)&sVt[(dh + d + 0) * 32 + kp2] = pack_bf16(va.x, vb.x);
                *(unsigned*)&sVt[(dh + d + 1) * 32 + kp2] = pack_bf16(va.y, vb.y);
                *(unsigned*)&sVt[(dh + d + 2) * 32 + kp2] = pack_bf16(va.z, vb.z);
                *(unsigned*)&sVt[(dh + d + 3) * 32 + kp2] = pack_bf16(va.w, vb.w);
            }
        }
        asm volatile("s_wait_dscnt 0x0" ::: "memory");

        bfvec16 bk[8];
#pragma unroll
        for (int c = 0; c < 4; ++c) {
            const int dimb = c * 32 + ((lane >> 4) << 4);
#pragma unroll
            for (int e = 0; e < 16; ++e) {
                bk[2 * c + 0][e] = bfbits(sK[ln16 * HEAD_DIM + dimb + e]);
                bk[2 * c + 1][e] = bfbits(sK[(16 + ln16) * HEAD_DIM + dimb + e]);
            }
        }
        __builtin_amdgcn_sched_barrier(0);
        fvec8 s0 = fvec8{0, 0, 0, 0, 0, 0, 0, 0};
        fvec8 s1 = fvec8{0, 0, 0, 0, 0, 0, 0, 0};
#pragma unroll
        for (int c = 0; c < 4; ++c) {
            s0 = wmma_bf16(qf[c], bk[2 * c + 0], s0);
            s1 = wmma_bf16(qf[c], bk[2 * c + 1], s1);
        }
        __builtin_amdgcn_sched_barrier(0);

        float mx[8];
#pragma unroll
        for (int r = 0; r < 8; ++r) {
            float a = s0[r], bb = s1[r];
            if (kbase + ln16 >= ctx)      a = NEG_INF;
            if (kbase + 16 + ln16 >= ctx) bb = NEG_INF;
            s0[r] = a; s1[r] = bb;
            mx[r] = fmaxf(a, bb);
        }
#pragma unroll
        for (int off = 1; off <= 8; off <<= 1)
#pragma unroll
            for (int r = 0; r < 8; ++r)
                mx[r] = fmaxf(mx[r], __shfl_xor(mx[r], off, 32));

        float alpha[8];
#pragma unroll
        for (int r = 0; r < 8; ++r) {
            const float mn = fmaxf(m_i[r], mx[r]);
            alpha[r] = __expf(m_i[r] - mn);
            m_i[r]   = mn;
            const float p0 = __expf(s0[r] - mn);
            const float p1 = __expf(s1[r] - mn);
            sP[(rowb + r) * 32 + ln16]      = f2bf(p0);
            sP[(rowb + r) * 32 + 16 + ln16] = f2bf(p1);
#pragma unroll
            for (int c = 0; c < 8; ++c) o[c][r] *= alpha[r];
        }
        asm volatile("s_wait_dscnt 0x0" ::: "memory");

        bfvec16 pf;
#pragma unroll
        for (int e = 0; e < 16; ++e)
            pf[e] = bfbits(sP[ln16 * 32 + a_kidx(lane, e)]);
        bfvec16 vf[8];
#pragma unroll
        for (int c = 0; c < 8; ++c) {
            const int dimc = c * 16 + ln16;
            const int keyb = (lane >> 4) << 4;
#pragma unroll
            for (int e = 0; e < 16; ++e)
                vf[c][e] = bfbits(sVt[dimc * 32 + keyb + e]);
        }
        __builtin_amdgcn_sched_barrier(0);
        fvec8 lf = wmma_bf16(pf, ones, fvec8{0, 0, 0, 0, 0, 0, 0, 0});
#pragma unroll
        for (int c = 0; c < 8; ++c)
            o[c] = wmma_bf16(pf, vf[c], o[c]);
        __builtin_amdgcn_sched_barrier(0);
#pragma unroll
        for (int r = 0; r < 8; ++r) l_i[r] = l_i[r] * alpha[r] + lf[r];
        asm volatile("s_wait_dscnt 0x0" ::: "memory");
    }

    if (lane < 16) {
#pragma unroll
        for (int c = 0; c < 8; ++c)
#pragma unroll
            for (int r = 0; r < 4; ++r) {
                const int head = kvh * GQA + r;
                out[((size_t)token * NUM_HEADS + head) * HEAD_DIM + c * 16 + ln16] =
                    o[c][r] / l_i[r];
            }
    }
}

// ---------------- launcher ----------------
extern "C" void kernel_launch(void* const* d_in, const int* in_sizes, int n_in,
                              void* d_out, int out_size, void* d_ws, size_t ws_size,
                              hipStream_t stream) {
    const float*     q            = (const float*)d_in[0];
    const float*     k            = (const float*)d_in[1];
    const float*     v            = (const float*)d_in[2];
    const float*     k_cache      = (const float*)d_in[3];
    const float*     v_cache      = (const float*)d_in[4];
    const long long* slot_mapping = (const long long*)d_in[5];
    const int*       context_lens = (const int*)d_in[6];
    const int*       block_tables = (const int*)d_in[7];
    float*           out          = (float*)d_out;
    int*             inv          = (int*)d_ws;   // 65536 ints = 256 KB

    init_inv_kernel<<<NUM_SLOTS / 256, 256, 0, stream>>>(inv);
    scatter_inv_kernel<<<(TOTAL + 255) / 256, 256, 0, stream>>>(slot_mapping, inv);

    prefill_fa_kernel<<<NUM_SEQS * NUM_HEADS * (SEQLEN / 64), 128, 0, stream>>>(
        q, k, v, out);

    decode_fa_kernel<<<DECODE_BATCH * NUM_KV_HEADS, 32, 0, stream>>>(
        q, k, v, k_cache, v_cache, inv, context_lens, block_tables, out);
}
